// TemporalTransformer_81604378624404
// MI455X (gfx1250) — matrix operand order repaired and checked by reading, not compiled, not run on hardware
//
#include <hip/hip_runtime.h>
#include <math.h>

typedef __attribute__((ext_vector_type(16))) _Float16 v16h;
typedef __attribute__((ext_vector_type(8)))  _Float16 v8h;
typedef __attribute__((ext_vector_type(8)))  float    v8f;

#define Bn   32
#define Sn   511
#define Tn   512
#define Dn   512
#define Hn   8
#define HDn  64
#define FDn  448
#define DFFn 2048

#define FLAG_BIAS 1
#define FLAG_RES  2
#define FLAG_ACC  4
#define FLAG_GELU 8

union AV { v16h v; v8h h2[2]; _Float16 e[16]; };

__device__ __forceinline__ float warp_sum32(float v) {
#pragma unroll
  for (int off = 16; off > 0; off >>= 1) v += __shfl_xor(v, off, 32);
  return v;
}
__device__ __forceinline__ float warp_max32(float v) {
#pragma unroll
  for (int off = 16; off > 0; off >>= 1) v = fmaxf(v, __shfl_xor(v, off, 32));
  return v;
}

// pack two fp32 into packed-f16 dword (lowers to v_cvt_pk_f16_f32)
__device__ __forceinline__ unsigned int pkh(float a, float b) {
  union { _Float16 h[2]; unsigned int u; } c;
  c.h[0] = (_Float16)a; c.h[1] = (_Float16)b;
  return c.u;
}

// pack a float4 pair into 8 f16 lanes of a fragment
__device__ __forceinline__ v8h cvt8(float4 a, float4 b) {
  v8h r;
  r[0] = (_Float16)a.x; r[1] = (_Float16)a.y; r[2] = (_Float16)a.z; r[3] = (_Float16)a.w;
  r[4] = (_Float16)b.x; r[5] = (_Float16)b.y; r[6] = (_Float16)b.z; r[7] = (_Float16)b.w;
  return r;
}

// ---------------------------------------------------------------------------
// Embedding + sinusoidal PE.
// ---------------------------------------------------------------------------
__global__ void embed_kernel(const float* __restrict__ runs,
                             const float* __restrict__ wick,
                             const float* __restrict__ overs,
                             const int*   __restrict__ batters,
                             const int*   __restrict__ bowlers,
                             const float* __restrict__ pemb,
                             const float* __restrict__ Wf,
                             const float* __restrict__ bf,
                             const float* __restrict__ qtok,
                             float* __restrict__ X)
{
  long idx = (long)blockIdx.x * blockDim.x + threadIdx.x;
  if (idx >= (long)Bn * Tn * Dn) return;
  int d = (int)(idx % Dn);
  long bt = idx / Dn;
  int t = (int)(bt % Tn);
  int b = (int)(bt / Tn);
  float val;
  if (t == Tn - 1) {
    val = qtok[d];
  } else {
    long s = (long)b * Sn + t;
    if (d < FDn) {
      val = runs[s] * Wf[d] + wick[s] * Wf[FDn + d] + overs[s] * Wf[2 * FDn + d] + bf[d];
    } else if (d < FDn + 32) {
      val = pemb[batters[s] * 32 + (d - FDn)];
    } else {
      val = pemb[bowlers[s] * 32 + (d - FDn - 32)];
    }
  }
  float freq = expf(-(float)(d & ~1) * (9.210340371976184f / (float)Dn));
  float ang = (float)t * freq;
  val += (d & 1) ? cosf(ang) : sinf(ang);
  X[idx] = val;
}

// ---------------------------------------------------------------------------
// GEMM via v_wmma_f32_16x16x32_f16 with LDS staging.
// Block = 256 threads (8 waves) computes a 128x64 tile of
//   Y = A(MxK fp32, lda) @ W(KxN fp32, ldw) [+bias][GELU][+R][+=Y].
// FULL=true means M is known tile-divisible (no row guards anywhere).
// ---------------------------------------------------------------------------
#define BM 128
#define BN 64

template <bool FULL>
__global__ __launch_bounds__(256)
void gemm_wmma(const float* __restrict__ A, long lda,
               const float* __restrict__ W, long ldw,
               const float* __restrict__ bias,
               const float* __restrict__ R, long ldr,
               float* __restrict__ Y, long ldy,
               int M, int N, int K, int flags)
{
  __shared__ _Float16 As[BM * 32];
  __shared__ _Float16 Bs[BN * 32];

  int tid  = threadIdx.x;
  int wave = tid >> 5;
  int lane = tid & 31;
  int half = lane >> 4;
  int mr   = lane & 15;
  int ntb  = N / BN;
  int bm0  = (blockIdx.x / ntb) * BM;
  int bn0  = (blockIdx.x % ntb) * BN;

  v8f acc[4];
#pragma unroll
  for (int j = 0; j < 4; ++j) acc[j] = (v8f){0.f,0.f,0.f,0.f,0.f,0.f,0.f,0.f};

  for (int k0 = 0; k0 < K; k0 += 32) {
    // ---- cooperative A tile: 128x32 fp32 -> f16 ----
#pragma unroll
    for (int i = 0; i < 4; ++i) {
      int slot = tid + i * 256;
      int row  = slot >> 3;
      int c4   = slot & 7;
      union { _Float16 h[4]; uint2 u; } cv;
      if (FULL || (bm0 + row < M)) {
        const float4 f = *(const float4*)(A + (long)(bm0 + row) * lda + k0 + c4 * 4);
        cv.h[0] = (_Float16)f.x; cv.h[1] = (_Float16)f.y;
        cv.h[2] = (_Float16)f.z; cv.h[3] = (_Float16)f.w;
      } else {
        cv.h[0] = cv.h[1] = cv.h[2] = cv.h[3] = (_Float16)0.f;
      }
      *(uint2*)&As[row * 32 + c4 * 4] = cv.u;
    }
    // ---- cooperative B tile: 32x64 fp32 -> f16 transposed [n][k] ----
    {
      int kp = tid >> 4;
      int c4 = tid & 15;
      const float* w0 = W + (long)(k0 + 2 * kp) * ldw + bn0 + c4 * 4;
      const float4 f0 = *(const float4*)(w0);
      const float4 f1 = *(const float4*)(w0 + ldw);
      *(unsigned int*)&Bs[(c4 * 4 + 0) * 32 + 2 * kp] = pkh(f0.x, f1.x);
      *(unsigned int*)&Bs[(c4 * 4 + 1) * 32 + 2 * kp] = pkh(f0.y, f1.y);
      *(unsigned int*)&Bs[(c4 * 4 + 2) * 32 + 2 * kp] = pkh(f0.z, f1.z);
      *(unsigned int*)&Bs[(c4 * 4 + 3) * 32 + 2 * kp] = pkh(f0.w, f1.w);
    }
    __syncthreads();

    const _Float16* arow = &As[(wave * 16 + mr) * 32];
    AV au;
    au.h2[0] = *(const v8h*)(arow + half * 8);
    au.h2[1] = *(const v8h*)(arow + 16 + half * 8);
#pragma unroll
    for (int j = 0; j < 4; ++j) {
      const _Float16* brow = &Bs[(j * 16 + mr) * 32];
      AV bu;
      bu.h2[0] = *(const v8h*)(brow + half * 8);
      bu.h2[1] = *(const v8h*)(brow + 16 + half * 8);
      acc[j] = __builtin_amdgcn_wmma_f32_16x16x32_f16(false, au.v, false, bu.v,
                                                      (short)0, acc[j], false, false);
    }
    __syncthreads();
  }

  // ---- epilogue ----
#pragma unroll
  for (int j = 0; j < 4; ++j) {
    int col = bn0 + j * 16 + mr;
    float bval = (flags & FLAG_BIAS) ? bias[col] : 0.f;
#pragma unroll
    for (int v = 0; v < 8; ++v) {
      int m = bm0 + wave * 16 + v + half * 8;
      if (FULL || (m < M)) {
        float val = acc[j][v] + bval;
        if (flags & FLAG_GELU) val = 0.5f * val * (1.f + erff(val * 0.70710678118654752f));
        if (flags & FLAG_RES)  val += R[(long)m * ldr + col];
        long yi = (long)m * ldy + col;
        if (flags & FLAG_ACC)  val += Y[yi];
        Y[yi] = val;
      }
    }
  }
}

// ---------------------------------------------------------------------------
// Fused attention: one wave per (b, h, 16-row q tile).
// Branchless head-bias masks: q-side ids preloaded to LDS, key-side ids
// loaded with clamped index (no exec-mask divergence in the hot loop).
// ---------------------------------------------------------------------------
__global__ __launch_bounds__(32)
void attn_wmma(const float* __restrict__ Q, const float* __restrict__ Km,
               const float* __restrict__ V, float* __restrict__ O,
               const int* __restrict__ batters, const int* __restrict__ bowlers,
               const float* __restrict__ recp, const float* __restrict__ bowp,
               const float* __restrict__ batp)
{
  __shared__ float    s[16][Tn];      // 32 KB
  __shared__ _Float16 pf[16][Tn];     // 16 KB
  __shared__ _Float16 Vt[HDn * 32];   //  4 KB, transposed [d][k]
  __shared__ int      qids[16];

  int g = blockIdx.x;
  int qt = g & 31; g >>= 5;
  int h  = g & 7;  int b = g >> 3;
  int q0 = qt << 4;
  int lane = threadIdx.x & 31;
  int half = lane >> 4;
  int mr   = lane & 15;

  float rs = recp[0];
  float hs = (h == 1) ? bowp[0] : batp[0];
  const int* ids = (h == 1) ? bowlers : ((h == 2) ? batters : nullptr);
  const long idbase = (long)b * Sn;

  // preload q-side actor ids (clamped index; row 511 handled by select later)
  if (ids && lane < 16) {
    int q = q0 + lane;
    qids[lane] = ids[idbase + (q < Sn ? q : Sn - 1)];
  }
  __syncthreads();

  const long baseQ = ((long)b * Tn + q0) * Dn + h * HDn;

  // hoisted Q fragments for both kc chunks
  AV auq[2];
#pragma unroll
  for (int c = 0; c < 2; ++c) {
    const float* qp = Q + baseQ + (long)mr * Dn + c * 32;
    auq[c].h2[0] = cvt8(*(const float4*)(qp + half * 8),
                        *(const float4*)(qp + half * 8 + 4));
    auq[c].h2[1] = cvt8(*(const float4*)(qp + 16 + half * 8),
                        *(const float4*)(qp + 16 + half * 8 + 4));
  }

  // ---- scores -> LDS ----
  for (int n0 = 0; n0 < Tn; n0 += 16) {
    v8f acc = {0.f,0.f,0.f,0.f,0.f,0.f,0.f,0.f};
    const long baseK = ((long)b * Tn + n0) * Dn + h * HDn;
#pragma unroll
    for (int c = 0; c < 2; ++c) {
      const float* kp = Km + baseK + (long)mr * Dn + c * 32;
      AV bu;
      bu.h2[0] = cvt8(*(const float4*)(kp + half * 8),
                      *(const float4*)(kp + half * 8 + 4));
      bu.h2[1] = cvt8(*(const float4*)(kp + 16 + half * 8),
                      *(const float4*)(kp + 16 + half * 8 + 4));
      acc = __builtin_amdgcn_wmma_f32_16x16x32_f16(false, auq[c].v, false, bu.v,
                                                   (short)0, acc, false, false);
    }
    int kpos = n0 + mr;
    int kid = 0;
    if (ids) kid = ids[idbase + (kpos < Sn ? kpos : Sn - 1)];  // wave-uniform branch
#pragma unroll
    for (int v = 0; v < 8; ++v) {
      int m = v + half * 8;
      int q = q0 + m;
      float sc = acc[v] * 0.125f;
      if (h == 0) {
        sc += rs * (float)kpos * (1.0f / (float)Tn);
      } else if (h == 1 || h == 2) {
        // branchless ones-padded actor mask
        int on = (int)(q == Tn - 1) | (int)(kpos == Tn - 1) | (int)(qids[m] == kid);
        sc += hs * (float)on;
      }
      sc = (kpos > q) ? -INFINITY : sc;
      s[m][kpos] = sc;
    }
  }
  __syncthreads();

  // ---- softmax per row ----
  for (int r = 0; r < 16; ++r) {
    float mx = -INFINITY;
    for (int j = lane; j < Tn; j += 32) mx = fmaxf(mx, s[r][j]);
    mx = warp_max32(mx);
    float sum = 0.f;
    for (int j = lane; j < Tn; j += 32) {
      float e = expf(s[r][j] - mx);
      s[r][j] = e;
      sum += e;
    }
    sum = warp_sum32(sum);
    float inv = 1.f / sum;
    for (int j = lane; j < Tn; j += 32) pf[r][j] = (_Float16)(s[r][j] * inv);
  }
  __syncthreads();

  // ---- O = P @ V ----
  v8f acc4[4];
#pragma unroll
  for (int j = 0; j < 4; ++j) acc4[j] = (v8f){0.f,0.f,0.f,0.f,0.f,0.f,0.f,0.f};

  for (int n0 = 0; n0 < Tn; n0 += 32) {
#pragma unroll
    for (int i = 0; i < 8; ++i) {
      int slot = lane + i * 32;
      int kp2  = slot >> 4;
      int c4   = slot & 15;
      const float* vp = V + ((long)b * Tn + n0 + 2 * kp2) * Dn + h * HDn + c4 * 4;
      const float4 f0 = *(const float4*)(vp);
      const float4 f1 = *(const float4*)(vp + Dn);
      *(unsigned int*)&Vt[(c4 * 4 + 0) * 32 + 2 * kp2] = pkh(f0.x, f1.x);
      *(unsigned int*)&Vt[(c4 * 4 + 1) * 32 + 2 * kp2] = pkh(f0.y, f1.y);
      *(unsigned int*)&Vt[(c4 * 4 + 2) * 32 + 2 * kp2] = pkh(f0.z, f1.z);
      *(unsigned int*)&Vt[(c4 * 4 + 3) * 32 + 2 * kp2] = pkh(f0.w, f1.w);
    }
    __syncthreads();
    AV au;
    const _Float16* prow = &pf[mr][n0];
    au.h2[0] = *(const v8h*)(prow + half * 8);
    au.h2[1] = *(const v8h*)(prow + 16 + half * 8);
#pragma unroll
    for (int dt = 0; dt < 4; ++dt) {
      const _Float16* vrow = &Vt[(dt * 16 + mr) * 32];
      AV bu;
      bu.h2[0] = *(const v8h*)(vrow + half * 8);
      bu.h2[1] = *(const v8h*)(vrow + 16 + half * 8);
      acc4[dt] = __builtin_amdgcn_wmma_f32_16x16x32_f16(false, au.v, false, bu.v,
                                                        (short)0, acc4[dt], false, false);
    }
    __syncthreads();
  }
#pragma unroll
  for (int dt = 0; dt < 4; ++dt) {
#pragma unroll
    for (int v = 0; v < 8; ++v) {
      int m = v + half * 8;
      O[((long)b * Tn + q0 + m) * Dn + h * HDn + dt * 16 + mr] = acc4[dt][v];
    }
  }
}

// ---------------------------------------------------------------------------
// LayerNorm: one wave per row of 512, float4 vectorized.
// ---------------------------------------------------------------------------
__global__ __launch_bounds__(256)
void ln_kernel(const float* __restrict__ in, const float* __restrict__ g,
               const float* __restrict__ be, float* __restrict__ out, int rows)
{
  int wave = threadIdx.x >> 5;
  int lane = threadIdx.x & 31;
  long r = (long)blockIdx.x * 8 + wave;
  if (r >= rows) return;
  const float* x = in + r * Dn;
  float4 f[4];
  float sum = 0.f;
#pragma unroll
  for (int i = 0; i < 4; ++i) {
    f[i] = *(const float4*)(x + lane * 4 + i * 128);
    sum += f[i].x + f[i].y + f[i].z + f[i].w;
  }
  float mu = warp_sum32(sum) * (1.f / (float)Dn);
  float vs = 0.f;
#pragma unroll
  for (int i = 0; i < 4; ++i) {
    float dx = f[i].x - mu, dy = f[i].y - mu, dz = f[i].z - mu, dw = f[i].w - mu;
    vs += dx * dx + dy * dy + dz * dz + dw * dw;
  }
  float var = warp_sum32(vs) * (1.f / (float)Dn);
  float inv = rsqrtf(var + 1e-5f);
#pragma unroll
  for (int i = 0; i < 4; ++i) {
    int c = lane * 4 + i * 128;
    float4 gg = *(const float4*)(g + c);
    float4 bb = *(const float4*)(be + c);
    float4 o;
    o.x = (f[i].x - mu) * inv * gg.x + bb.x;
    o.y = (f[i].y - mu) * inv * gg.y + bb.y;
    o.z = (f[i].z - mu) * inv * gg.z + bb.z;
    o.w = (f[i].w - mu) * inv * gg.w + bb.w;
    *(float4*)(out + r * Dn + c) = o;
  }
}

// ---------------------------------------------------------------------------
extern "C" void kernel_launch(void* const* d_in, const int* in_sizes, int n_in,
                              void* d_out, int out_size, void* d_ws, size_t ws_size,
                              hipStream_t stream)
{
  (void)in_sizes; (void)n_in; (void)out_size; (void)ws_size;
  const float* runs    = (const float*)d_in[0];
  const float* wick    = (const float*)d_in[1];
  const float* overs   = (const float*)d_in[2];
  const int*   batters = (const int*)d_in[3];
  const int*   bowlers = (const int*)d_in[4];
  const float* pemb    = (const float*)d_in[5];
  const float* Wf      = (const float*)d_in[6];
  const float* bf      = (const float*)d_in[7];
  const float* qtok    = (const float*)d_in[8];
  const float* Wq      = (const float*)d_in[9];
  const float* bq      = (const float*)d_in[10];
  const float* Wk      = (const float*)d_in[11];
  const float* bk      = (const float*)d_in[12];
  const float* Wv      = (const float*)d_in[13];
  const float* bv      = (const float*)d_in[14];
  const float* Wo      = (const float*)d_in[15];
  const float* bo      = (const float*)d_in[16];
  const float* rec_s   = (const float*)d_in[17];
  const float* bow_s   = (const float*)d_in[18];
  const float* bat_s   = (const float*)d_in[19];
  const float* W1      = (const float*)d_in[20];
  const float* b1      = (const float*)d_in[21];
  const float* W2      = (const float*)d_in[22];
  const float* b2      = (const float*)d_in[23];
  const float* g1      = (const float*)d_in[24];
  const float* be1     = (const float*)d_in[25];
  const float* g2      = (const float*)d_in[26];
  const float* be2     = (const float*)d_in[27];
  const float* Wout    = (const float*)d_in[28];
  const float* bout    = (const float*)d_in[29];

  const int NTOK = Bn * Tn;                    // 16384 rows
  const size_t BUF = (size_t)NTOK * Dn;        // 8M floats = 32 MB
  float* X  = (float*)d_ws;
  float* Qb = X + BUF;
  float* Kb = X + 2 * BUF;
  float* Vb = X + 3 * BUF;
  float* Ob = X + 4 * BUF;
  float* T1 = X + 5 * BUF;

  {
    long tot = (long)Bn * Tn * Dn;
    embed_kernel<<<(unsigned)((tot + 255) / 256), 256, 0, stream>>>(
        runs, wick, overs, batters, bowlers, pemb, Wf, bf, qtok, X);
  }

  const int gemmBlk = (NTOK / BM) * (Dn / BN);  // 128 * 8 = 1024

  for (int l = 0; l < 4; ++l) {
    const float* wq = Wq + (size_t)l * Dn * Dn;
    const float* wk = Wk + (size_t)l * Dn * Dn;
    const float* wv = Wv + (size_t)l * Dn * Dn;
    const float* wo = Wo + (size_t)l * Dn * Dn;

    gemm_wmma<true><<<gemmBlk, 256, 0, stream>>>(X, Dn, wq, Dn, bq + l * Dn,
                                                 nullptr, 0, Qb, Dn, NTOK, Dn, Dn, FLAG_BIAS);
    gemm_wmma<true><<<gemmBlk, 256, 0, stream>>>(X, Dn, wk, Dn, bk + l * Dn,
                                                 nullptr, 0, Kb, Dn, NTOK, Dn, Dn, FLAG_BIAS);
    gemm_wmma<true><<<gemmBlk, 256, 0, stream>>>(X, Dn, wv, Dn, bv + l * Dn,
                                                 nullptr, 0, Vb, Dn, NTOK, Dn, Dn, FLAG_BIAS);

    attn_wmma<<<Bn * Hn * (Tn / 16), 32, 0, stream>>>(
        Qb, Kb, Vb, Ob, batters, bowlers, rec_s + l, bow_s + l, bat_s + l);

    gemm_wmma<true><<<gemmBlk, 256, 0, stream>>>(Ob, Dn, wo, Dn, bo + l * Dn,
                                                 X, Dn, T1, Dn, NTOK, Dn, Dn,
                                                 FLAG_BIAS | FLAG_RES);
    ln_kernel<<<NTOK / 8, 256, 0, stream>>>(T1, g1 + l * Dn, be1 + l * Dn, X, NTOK);

    for (int c = 0; c < 4; ++c) {
      gemm_wmma<true><<<gemmBlk, 256, 0, stream>>>(
          X, Dn,
          W1 + (size_t)l * Dn * DFFn + (size_t)c * 512, DFFn,
          b1 + (size_t)l * DFFn + (size_t)c * 512,
          nullptr, 0, T1, 512, NTOK, 512, Dn, FLAG_BIAS | FLAG_GELU);
      int f = (c == 0) ? (FLAG_BIAS | FLAG_RES) : FLAG_ACC;
      gemm_wmma<true><<<gemmBlk, 256, 0, stream>>>(
          T1, 512,
          W2 + (size_t)l * DFFn * Dn + (size_t)c * 512 * Dn, Dn,
          b2 + l * Dn, X, Dn, Ob, Dn, NTOK, Dn, 512, f);
    }
    ln_kernel<<<NTOK / 8, 256, 0, stream>>>(Ob, g2 + l * Dn, be2 + l * Dn, X, NTOK);
  }

  gemm_wmma<false><<<Dn / BN, 256, 0, stream>>>(X + (size_t)(Tn - 1) * Dn, (long)Tn * Dn,
                                                Wout, Dn, bout, nullptr, 0,
                                                (float*)d_out, Dn, Bn, Dn, Dn, FLAG_BIAS);
}
